// Net_89223650607597
// MI455X (gfx1250) — matrix operand order repaired and checked by reading, not compiled, not run on hardware
//
#include <hip/hip_runtime.h>
#include <math.h>

// ---------------------------------------------------------------------------
// CDNA5 (gfx1250) wave32 WMMA pipeline for LeNet + 2-expert MoE head.
// All GEMM-shaped work (conv1, conv2, fc1, fc2, experts, fc4) runs on
// v_wmma_f32_16x16x32_f16 with f32 accumulate.
// Activations: row-major [b][K_pad] f16. Weights: prepped to [n][K_pad] f16.
// Fragment layouts follow CDNA5 ISA 7.12.2 exactly:
//   A (16x32 f16): lane m=lane&15, sel=lane>>4; halves at k0+sel*8 and k0+sel*8+16
//   B (32x16 f16): lane n=lane&15, sel=lane>>4; halves at k0+sel*8 and k0+sel*8+16
//   C/D (16x16 f32): lane n=lane&15; VGPR r -> row m = r + 8*(lane>>4)
// ---------------------------------------------------------------------------

typedef __attribute__((ext_vector_type(16))) _Float16 v16h;
typedef __attribute__((ext_vector_type(8)))  _Float16 v8h;
typedef __attribute__((ext_vector_type(8)))  float    v8f;

union Frag16 { v16h v; v8h h[2]; };

__device__ __forceinline__ v16h load_a_frag(const _Float16* __restrict__ X, int ldk,
                                            int m0, int k0, int lane) {
  const int sel = lane >> 4, mr = lane & 15;
  const _Float16* p = X + (size_t)(m0 + mr) * ldk + k0 + sel * 8;
  Frag16 f;
  f.h[0] = *(const v8h*)(p);        // VGPRs 0-3: K = k0 + sel*8 .. +7
  f.h[1] = *(const v8h*)(p + 16);   // VGPRs 4-7: K = k0+16+sel*8 .. +7
  return f.v;
}

__device__ __forceinline__ v16h load_b_frag(const _Float16* __restrict__ W, int ldk,
                                            int n0, int k0, int lane) {
  const int sel = lane >> 4, nr = lane & 15;
  const _Float16* p = W + (size_t)(n0 + nr) * ldk + k0 + sel * 8;
  Frag16 f;
  f.h[0] = *(const v8h*)(p);        // K = k0+sel*8 .. +7
  f.h[1] = *(const v8h*)(p + 16);   // K = k0+16+sel*8 .. +7
  return f.v;
}

#define WMMA_F16(A, B, C) \
  __builtin_amdgcn_wmma_f32_16x16x32_f16(false, (A), false, (B), (short)0, (C), false, false)

// ---------------------------------------------------------------------------
// Prep kernels
// ---------------------------------------------------------------------------
__global__ void cvt_pad_f16_kernel(const float* __restrict__ src, _Float16* __restrict__ dst,
                                   int srcR, int srcC, int dstR, int dstC, int ldDst,
                                   int transpose) {
  int i = blockIdx.x * blockDim.x + threadIdx.x;
  if (i >= dstR * dstC) return;
  int r = i / dstC, c = i % dstC;
  float v = 0.f;
  if (transpose) { if (c < srcR && r < srcC) v = src[(size_t)c * srcC + r]; }
  else           { if (r < srcR && c < srcC) v = src[(size_t)r * srcC + c]; }
  dst[(size_t)r * ldDst + c] = (_Float16)v;
}

__global__ void pad_f32_kernel(const float* __restrict__ src, float* __restrict__ dst,
                               int srcN, int dstN) {
  int i = blockIdx.x * blockDim.x + threadIdx.x;
  if (i >= dstN) return;
  dst[i] = (i < srcN) ? src[i] : 0.f;
}

// conv1 weights -> [16][128] f16, K mapping k = (ci*5+kr)*8 + kc.
// Zeros at n>=6, tap-row>=15, kc>=5 guarantee pad lanes of A contribute 0.
__global__ void prep_conv1_w_kernel(const float* __restrict__ w, _Float16* __restrict__ dst) {
  int i = blockIdx.x * blockDim.x + threadIdx.x;
  if (i >= 16 * 128) return;
  int n = i / 128, k = i % 128;
  int r = k / 8, kc = k % 8;
  float v = 0.f;
  if (n < 6 && r < 15 && kc < 5) {
    int ci = r / 5, kr = r % 5;
    v = w[((n * 3 + ci) * 5 + kr) * 5 + kc];
  }
  dst[i] = (_Float16)v;
}

// ---------------------------------------------------------------------------
// conv1 (3->6ch, 5x5) + ReLU + 2x2 maxpool as WMMA implicit GEMM.
// One wave = (16-batch tile, one POOLED position). M=16 batches, N=16 (6 valid),
// K=(ci,kr) tap-rows padded to 8 kc each -> 128 = 4 chunks; 4 conv taps pooled
// in registers -> 16 WMMAs per wave.
// Dual-parity LDS staging: the 6-wide raw patch is stored twice (dc=0 and dc=1
// column shift) so every A-fragment half is a 16B-aligned ds_load_b128.
// Output h1: [b][6][14][14] f16.
// ---------------------------------------------------------------------------
__global__ __launch_bounds__(128) void conv1_wmma_kernel(const float* __restrict__ x,
                                                         const _Float16* __restrict__ W,
                                                         const float* __restrict__ bias16,
                                                         _Float16* __restrict__ h1, int B) {
  // [wave][par][bl][ci][xr][8] halves : 4 * 2*16*3*6*8 = 4 * 4608
  __shared__ __align__(16) _Float16 smem[4 * 4608];  // 36,864 B
  const int wv = threadIdx.x >> 5, lane = threadIdx.x & 31;
  const int gwave = blockIdx.x * 4 + wv;
  const int totalWaves = (B / 16) * 196;
  if (gwave >= totalWaves) return;
  const int pos = gwave % 196, mt = gwave / 196;   // mt-major => x stays L2-hot
  const int pr = pos / 14, pc = pos % 14;
  const int m0 = mt * 16;
  _Float16* ls = smem + wv * 4608;

  // stage 288 rows (bl, ci, xr): 6 f32 -> two shifted 8-half rows (pads zeroed)
  for (int row = lane; row < 288; row += 32) {
    int bl = row / 18, rem = row % 18;
    int ci = rem / 6, xr = rem % 6;
    const float* src = x + (((size_t)(m0 + bl) * 3 + ci) * 32 + (2 * pr + xr)) * 32 + 2 * pc;
    _Float16 v[6];
    #pragma unroll
    for (int j = 0; j < 6; ++j) v[j] = (_Float16)src[j];
    _Float16* d0 = ls + ((0 * 16 + bl) * 3 + ci) * 48 + xr * 8;
    _Float16* d1 = ls + ((1 * 16 + bl) * 3 + ci) * 48 + xr * 8;
    #pragma unroll
    for (int j = 0; j < 8; ++j) d0[j] = (j < 5) ? v[j] : (_Float16)0.f;
    #pragma unroll
    for (int j = 0; j < 8; ++j) d1[j] = (j < 5) ? v[j + 1] : (_Float16)0.f;
  }
  __syncthreads();

  const int sel = lane >> 4, bl = lane & 15;
  const float bv = bias16[lane & 15];
  v8f acc[2][2];
  #pragma unroll
  for (int dr = 0; dr < 2; ++dr)
    #pragma unroll
    for (int dc = 0; dc < 2; ++dc)
      #pragma unroll
      for (int i = 0; i < 8; ++i) acc[dr][dc][i] = bv;

  #pragma unroll
  for (int kk = 0; kk < 4; ++kk) {
    v16h bfr = load_b_frag(W, 128, 0, kk * 32, lane);   // shared by all 4 taps
    // tap-row indices for the two fragment halves (clamped; row 15 has 0 weights)
    int r1 = kk * 4 + sel;
    int r2 = r1 + 2;
    int rr1 = r1 > 14 ? 14 : r1;
    int rr2 = r2 > 14 ? 14 : r2;
    int ci1 = rr1 / 5, kr1 = rr1 % 5;
    int ci2 = rr2 / 5, kr2 = rr2 % 5;
    #pragma unroll
    for (int dr = 0; dr < 2; ++dr)
      #pragma unroll
      for (int dc = 0; dc < 2; ++dc) {
        const _Float16* p1 = ls + ((dc * 16 + bl) * 3 + ci1) * 48 + (dr + kr1) * 8;
        const _Float16* p2 = ls + ((dc * 16 + bl) * 3 + ci2) * 48 + (dr + kr2) * 8;
        Frag16 f;
        f.h[0] = *(const v8h*)p1;
        f.h[1] = *(const v8h*)p2;
        acc[dr][dc] = WMMA_F16(f.v, bfr, acc[dr][dc]);
      }
  }

  const int n = lane & 15;
  if (n < 6) {
    #pragma unroll
    for (int i = 0; i < 8; ++i) {
      float m = fmaxf(fmaxf(acc[0][0][i], acc[0][1][i]), fmaxf(acc[1][0][i], acc[1][1][i]));
      m = fmaxf(m, 0.f);
      int mm = sel * 8 + i;
      h1[(((size_t)(m0 + mm) * 6 + n) * 14 + pr) * 14 + pc] = (_Float16)m;
    }
  }
}

// ---------------------------------------------------------------------------
// conv2 as implicit GEMM on WMMA. One wave = (16-batch tile, one output pos).
// M=16 batches, N=16 out-channels (exact), K=6*5*5=150 padded to 160.
// im2col patch staged into wave-private LDS; 5 WMMAs; bias in C-init; ReLU fused.
// y2: [b][16][10][10] f16 (post-ReLU, pre-pool).
// ---------------------------------------------------------------------------
__global__ __launch_bounds__(128) void conv2_kernel(const _Float16* __restrict__ h1,
                                                    const _Float16* __restrict__ W,
                                                    const float* __restrict__ bias,
                                                    _Float16* __restrict__ y2, int B) {
  __shared__ __align__(16) _Float16 smem[4 * 16 * 160];  // 20 KB, 5 KB per wave
  const int wv = threadIdx.x >> 5, lane = threadIdx.x & 31;
  const int gwave = blockIdx.x * 4 + wv;
  const int totalWaves = (B / 16) * 100;
  if (gwave >= totalWaves) return;
  const int pos = gwave % 100, mt = gwave / 100;   // mt-major => h1 tile L2-hot
  const int r = pos / 10, c = pos % 10;
  const int m0 = mt * 16;
  _Float16* ls = smem + wv * (16 * 160);

  for (int row = lane; row < 480; row += 32) {
    int bl = row / 30, rem = row % 30;
    int ci = rem / 5, kr = rem % 5;
    const _Float16* src = h1 + (((size_t)(m0 + bl) * 6 + ci) * 14 + (r + kr)) * 14 + c;
    _Float16* dst = ls + bl * 160 + ci * 25 + kr * 5;
    #pragma unroll
    for (int kc = 0; kc < 5; ++kc) dst[kc] = src[kc];
  }
  for (int i = lane; i < 160; i += 32)           // zero K-pad 150..159
    ls[(i / 10) * 160 + 150 + (i % 10)] = (_Float16)0.f;
  __syncthreads();

  v8f acc;
  const float bv = bias[lane & 15];              // N = out channel
  #pragma unroll
  for (int i = 0; i < 8; ++i) acc[i] = bv;
  #pragma unroll
  for (int kk = 0; kk < 5; ++kk) {
    v16h a = load_a_frag(ls, 160, 0, kk * 32, lane);
    v16h b = load_b_frag(W, 160, 0, kk * 32, lane);
    acc = WMMA_F16(a, b, acc);
  }
  const int n = lane & 15, sel = lane >> 4;
  #pragma unroll
  for (int i = 0; i < 8; ++i) {
    int m = sel * 8 + i;
    y2[((size_t)(m0 + m) * 16 + n) * 100 + pos] = (_Float16)fmaxf(acc[i], 0.f);
  }
}

// 2x2 maxpool over y2, scatter into flattened fc1 layout h2[b][416] (ch*25+r*5+c).
__global__ void pool2_kernel(const _Float16* __restrict__ y2, _Float16* __restrict__ h2,
                             int B) {
  int gid = blockIdx.x * blockDim.x + threadIdx.x;
  int total = B * 16 * 25;
  if (gid >= total) return;
  int pc = gid % 5; int t = gid / 5;
  int pr = t % 5;   t /= 5;
  int ch = t % 16;  int b = t / 16;
  const _Float16* p = y2 + ((size_t)b * 16 + ch) * 100 + (2 * pr) * 10 + 2 * pc;
  float m = fmaxf(fmaxf((float)p[0], (float)p[1]), fmaxf((float)p[10], (float)p[11]));
  h2[(size_t)b * 416 + ch * 25 + pr * 5 + pc] = (_Float16)m;
}

// ---------------------------------------------------------------------------
// Generic ReLU(X @ W^T + b) on WMMA. One wave = 16-batch tile x full N strip.
// ---------------------------------------------------------------------------
template <int LDX, int NT, int KC, int LDY>
__global__ __launch_bounds__(128) void fc_relu_kernel(const _Float16* __restrict__ X,
                                                      const _Float16* __restrict__ W,
                                                      const float* __restrict__ bias,
                                                      _Float16* __restrict__ Y, int B) {
  const int wv = threadIdx.x >> 5, lane = threadIdx.x & 31;
  const int mt = blockIdx.x * 4 + wv;
  if (mt >= B / 16) return;
  const int m0 = mt * 16;
  const int n = lane & 15, sel = lane >> 4;

  v8f acc[NT];
  #pragma unroll
  for (int nt = 0; nt < NT; ++nt) {
    float bv = bias[nt * 16 + n];
    #pragma unroll
    for (int i = 0; i < 8; ++i) acc[nt][i] = bv;
  }
  for (int kk = 0; kk < KC; ++kk) {
    v16h a = load_a_frag(X, LDX, m0, kk * 32, lane);   // shared across NT tiles
    #pragma unroll
    for (int nt = 0; nt < NT; ++nt) {
      v16h b = load_b_frag(W, LDX, nt * 16, kk * 32, lane);
      acc[nt] = WMMA_F16(a, b, acc[nt]);
    }
  }
  #pragma unroll
  for (int nt = 0; nt < NT; ++nt)
    #pragma unroll
    for (int i = 0; i < 8; ++i)
      Y[(size_t)(m0 + sel * 8 + i) * LDY + nt * 16 + n] =
          (_Float16)fmaxf(acc[nt][i], 0.f);
}

// ---------------------------------------------------------------------------
// MoE head: gate softmax (2 experts, VALU + lane shuffle), two 84->84 expert
// GEMMs on WMMA, prob-weighted combine in registers, LDS repack, fc4 on WMMA.
// ---------------------------------------------------------------------------
__global__ __launch_bounds__(128) void moe_kernel(const _Float16* __restrict__ X,  // h4 [B][96]
                                                  const float* __restrict__ gate_w, // [84][2]
                                                  const _Float16* __restrict__ expW, // [2][96][96]
                                                  const float* __restrict__ expB,    // [2][96]
                                                  const _Float16* __restrict__ fc4W, // [16][96]
                                                  const float* __restrict__ fc4B,    // [16]
                                                  float* __restrict__ out, int B) {
  __shared__ __align__(16) _Float16 csm[4 * 16 * 96];  // 12 KB
  const int wv = threadIdx.x >> 5, lane = threadIdx.x & 31;
  const int mt = blockIdx.x * 4 + wv;
  if (mt >= B / 16) return;
  const int m0 = mt * 16;
  const int nn = lane & 15, sel = lane >> 4;

  // gate logits: lane handles (batch = m0+nn, expert = sel)
  float logit = 0.f;
  {
    const _Float16* hp = X + (size_t)(m0 + nn) * 96;
    for (int k = 0; k < 84; ++k)
      logit = fmaf((float)hp[k], gate_w[k * 2 + sel], logit);
  }
  float other = __shfl_xor(logit, 16, 32);
  float pSelf = 1.f / (1.f + expf(other - logit));   // 2-way softmax, own expert
  // broadcast p(expert0) for tile rows m = sel*8 + i (held in lanes 0..15)
  float p0m[8], p1m[8];
  #pragma unroll
  for (int i = 0; i < 8; ++i) {
    p0m[i] = __shfl(pSelf, sel * 8 + i, 32);
    p1m[i] = 1.f - p0m[i];
  }

  // hoist A fragments of h4 (K = 96 -> 3 chunks)
  v16h afr[3];
  #pragma unroll
  for (int kk = 0; kk < 3; ++kk) afr[kk] = load_a_frag(X, 96, m0, kk * 32, lane);

  float comb[6][8];
  v8f acc[6];
  // expert 0
  #pragma unroll
  for (int nt = 0; nt < 6; ++nt) {
    float bv = expB[nt * 16 + nn];
    #pragma unroll
    for (int i = 0; i < 8; ++i) acc[nt][i] = bv;
  }
  #pragma unroll
  for (int kk = 0; kk < 3; ++kk)
    #pragma unroll
    for (int nt = 0; nt < 6; ++nt) {
      v16h b = load_b_frag(expW, 96, nt * 16, kk * 32, lane);
      acc[nt] = WMMA_F16(afr[kk], b, acc[nt]);
    }
  #pragma unroll
  for (int nt = 0; nt < 6; ++nt)
    #pragma unroll
    for (int i = 0; i < 8; ++i) comb[nt][i] = p0m[i] * acc[nt][i];
  // expert 1
  #pragma unroll
  for (int nt = 0; nt < 6; ++nt) {
    float bv = expB[96 + nt * 16 + nn];
    #pragma unroll
    for (int i = 0; i < 8; ++i) acc[nt][i] = bv;
  }
  #pragma unroll
  for (int kk = 0; kk < 3; ++kk)
    #pragma unroll
    for (int nt = 0; nt < 6; ++nt) {
      v16h b = load_b_frag(expW + 96 * 96, 96, nt * 16, kk * 32, lane);
      acc[nt] = WMMA_F16(afr[kk], b, acc[nt]);
    }
  #pragma unroll
  for (int nt = 0; nt < 6; ++nt)
    #pragma unroll
    for (int i = 0; i < 8; ++i) comb[nt][i] = fmaf(p1m[i], acc[nt][i], comb[nt][i]);

  // repack combined activations -> LDS as f16 A-operand for fc4
  _Float16* ls = csm + wv * (16 * 96);
  #pragma unroll
  for (int nt = 0; nt < 6; ++nt)
    #pragma unroll
    for (int i = 0; i < 8; ++i)
      ls[(size_t)(sel * 8 + i) * 96 + nt * 16 + nn] = (_Float16)comb[nt][i];
  __syncthreads();

  v8f accf;
  {
    float bv = fc4B[nn];
    #pragma unroll
    for (int i = 0; i < 8; ++i) accf[i] = bv;
  }
  #pragma unroll
  for (int kk = 0; kk < 3; ++kk) {
    v16h a = load_a_frag(ls, 96, 0, kk * 32, lane);
    v16h b = load_b_frag(fc4W, 96, 0, kk * 32, lane);
    accf = WMMA_F16(a, b, accf);
  }
  if (nn < 10) {
    #pragma unroll
    for (int i = 0; i < 8; ++i)
      out[(size_t)(m0 + sel * 8 + i) * 10 + nn] = accf[i];
  }
}

// ---------------------------------------------------------------------------
// Host-side orchestration
// ---------------------------------------------------------------------------
static inline int cdiv(long long a, long long b) { return (int)((a + b - 1) / b); }

extern "C" void kernel_launch(void* const* d_in, const int* in_sizes, int n_in,
                              void* d_out, int out_size, void* d_ws, size_t ws_size,
                              hipStream_t stream) {
  const float* x        = (const float*)d_in[0];
  const float* conv1_w  = (const float*)d_in[1];
  const float* conv1_b  = (const float*)d_in[2];
  const float* conv2_w  = (const float*)d_in[3];
  const float* conv2_b  = (const float*)d_in[4];
  const float* fc1_w    = (const float*)d_in[5];
  const float* fc1_b    = (const float*)d_in[6];
  const float* fc2_w    = (const float*)d_in[7];
  const float* fc2_b    = (const float*)d_in[8];
  const float* gate_w   = (const float*)d_in[9];
  const float* expert_w = (const float*)d_in[10];
  const float* expert_b = (const float*)d_in[11];
  const float* fc4_w    = (const float*)d_in[12];
  const float* fc4_b    = (const float*)d_in[13];
  float* out = (float*)d_out;
  const int B = in_sizes[0] / (3 * 32 * 32);

  char* ws = (char*)d_ws;
  size_t off = 0;
  auto alloc = [&](size_t bytes) -> void* {
    void* p = ws + off;
    off = (off + bytes + 255) & ~(size_t)255;
    return p;
  };
  _Float16* h1   = (_Float16*)alloc((size_t)B * 6 * 196 * sizeof(_Float16));
  _Float16* y2   = (_Float16*)alloc((size_t)B * 16 * 100 * sizeof(_Float16));
  _Float16* h2   = (_Float16*)alloc((size_t)B * 416 * sizeof(_Float16));
  _Float16* h3   = (_Float16*)alloc((size_t)B * 128 * sizeof(_Float16));
  _Float16* h4   = (_Float16*)alloc((size_t)B * 96 * sizeof(_Float16));
  _Float16* w1c  = (_Float16*)alloc(16 * 128 * sizeof(_Float16));
  _Float16* w2h  = (_Float16*)alloc(16 * 160 * sizeof(_Float16));
  _Float16* w1h  = (_Float16*)alloc(128 * 416 * sizeof(_Float16));
  _Float16* w2fh = (_Float16*)alloc(96 * 128 * sizeof(_Float16));
  _Float16* weh  = (_Float16*)alloc(2 * 96 * 96 * sizeof(_Float16));
  _Float16* w4h  = (_Float16*)alloc(16 * 96 * sizeof(_Float16));
  float* b1c  = (float*)alloc(16 * sizeof(float));
  float* b1p  = (float*)alloc(128 * sizeof(float));
  float* b2p  = (float*)alloc(96 * sizeof(float));
  float* bep  = (float*)alloc(2 * 96 * sizeof(float));
  float* b4p  = (float*)alloc(16 * sizeof(float));
  (void)ws_size; (void)n_in; (void)out_size;

  const int TB = 256;
  // --- weight/bias prep (runs every call; deterministic) ---
  prep_conv1_w_kernel<<<cdiv(16 * 128, TB), TB, 0, stream>>>(conv1_w, w1c);
  cvt_pad_f16_kernel<<<cdiv(16 * 160, TB), TB, 0, stream>>>(conv2_w, w2h, 16, 150, 16, 160, 160, 0);
  cvt_pad_f16_kernel<<<cdiv(128 * 416, TB), TB, 0, stream>>>(fc1_w, w1h, 120, 400, 128, 416, 416, 0);
  cvt_pad_f16_kernel<<<cdiv(96 * 128, TB), TB, 0, stream>>>(fc2_w, w2fh, 84, 120, 96, 128, 128, 0);
  cvt_pad_f16_kernel<<<cdiv(96 * 96, TB), TB, 0, stream>>>(expert_w, weh, 84, 84, 96, 96, 96, 1);
  cvt_pad_f16_kernel<<<cdiv(96 * 96, TB), TB, 0, stream>>>(expert_w + 84 * 84, weh + 96 * 96, 84, 84, 96, 96, 96, 1);
  cvt_pad_f16_kernel<<<cdiv(16 * 96, TB), TB, 0, stream>>>(fc4_w, w4h, 10, 84, 16, 96, 96, 0);
  // zero-fill h2 pad columns 400..415 (srcR=0 => pure zero fill)
  cvt_pad_f16_kernel<<<cdiv((long long)B * 16, TB), TB, 0, stream>>>(fc1_w, h2 + 400, 0, 0, B, 16, 416, 0);
  pad_f32_kernel<<<1, 16, 0, stream>>>(conv1_b, b1c, 6, 16);
  pad_f32_kernel<<<1, 128, 0, stream>>>(fc1_b, b1p, 120, 128);
  pad_f32_kernel<<<1, 96, 0, stream>>>(fc2_b, b2p, 84, 96);
  pad_f32_kernel<<<1, 96, 0, stream>>>(expert_b, bep, 84, 96);
  pad_f32_kernel<<<1, 96, 0, stream>>>(expert_b + 84, bep + 96, 84, 96);
  pad_f32_kernel<<<1, 16, 0, stream>>>(fc4_b, b4p, 10, 16);

  // --- conv1 (WMMA implicit GEMM, fused ReLU + maxpool) ---
  {
    long long waves = (long long)(B / 16) * 196;
    conv1_wmma_kernel<<<cdiv(waves, 4), 128, 0, stream>>>(x, w1c, b1c, h1, B);
  }
  // --- conv2 (WMMA implicit GEMM) ---
  {
    long long waves = (long long)(B / 16) * 100;
    conv2_kernel<<<cdiv(waves, 4), 128, 0, stream>>>(h1, w2h, conv2_b, y2, B);
  }
  // --- pool2 -> flattened fc1 input ---
  pool2_kernel<<<cdiv((long long)B * 16 * 25, TB), TB, 0, stream>>>(y2, h2, B);
  // --- fc1: [B,416] x [128,416]^T ---
  fc_relu_kernel<416, 8, 13, 128><<<cdiv(B / 16, 4), 128, 0, stream>>>(h2, w1h, b1p, h3, B);
  // --- fc2: [B,128] x [96,128]^T ---
  fc_relu_kernel<128, 6, 4, 96><<<cdiv(B / 16, 4), 128, 0, stream>>>(h3, w2fh, b2p, h4, B);
  // --- MoE + fc4 ---
  moe_kernel<<<cdiv(B / 16, 4), 128, 0, stream>>>(h4, gate_w, weh, bep, w4h, b4p, out, B);
}
